// MPNEncoder_57346403336640
// MI455X (gfx1250) — matrix operand order repaired and checked by reading, not compiled, not run on hardware
//
#include <hip/hip_runtime.h>
#include <hip/hip_bf16.h>

// ---------------------------------------------------------------------------
// Problem constants (from reference): N atoms, K neighbors, bond feats, etc.
// ---------------------------------------------------------------------------
#define NN      100000
#define KNEI    6
#define AFEAT   133
#define BFEAT   14
#define HH      256
#define DEPTH   3

typedef __attribute__((ext_vector_type(16))) __bf16        v16bf;
typedef __attribute__((ext_vector_type(8)))  float         v8f;
typedef __attribute__((ext_vector_type(8)))  unsigned int  v8u;

// float -> bf16 bits, round-to-nearest-even
__device__ __forceinline__ unsigned short f2bf(float f) {
    unsigned int x = __builtin_bit_cast(unsigned int, f);
    unsigned int r = x + 0x7FFFu + ((x >> 16) & 1u);
    return (unsigned short)(r >> 16);
}
__device__ __forceinline__ unsigned int pk2(float a, float b) {
    return (unsigned int)f2bf(a) | ((unsigned int)f2bf(b) << 16);
}

// ---------------------------------------------------------------------------
// Pack W[Ksrc x 256] (row-major, fp32) into bf16 WMMA-B fragment order.
// Packed element index: ((nt*nkc + kc)*32 + lane)*16 + e
//   element e of lane -> K = kc*32 + (e&7) + (e>=8 ? 16 : 0) + (lane>=16 ? 8 : 0), Ncol = nt*16 + lane%16
// Rows k >= Ksrc are zero (K padding).
// ---------------------------------------------------------------------------
__global__ void pack_w_kernel(const float* __restrict__ W, int Ksrc, int nkc,
                              unsigned short* __restrict__ dst) {
    int idx = blockIdx.x * 256 + threadIdx.x;           // < nkc*16*512 exactly
    int e     = idx & 15;
    int lane  = (idx >> 4) & 31;
    int chunk = idx >> 9;                               // nt*nkc + kc
    int kc = chunk % nkc;
    int nt = chunk / nkc;
    int k = kc * 32 + (e & 7) + ((e >> 3) << 4) + ((lane >> 4) ? 8 : 0);
    int n = nt * 16 + (lane & 15);
    float v = (k < Ksrc) ? W[(size_t)k * HH + n] : 0.0f;
    dst[idx] = f2bf(v);
}

// Pad f_atoms [N x 133] -> [N x 160] with zero columns.
__global__ void pad_atoms_kernel(const float* __restrict__ fa, float* __restrict__ dst) {
    size_t idx = (size_t)blockIdx.x * 256 + threadIdx.x;  // N*160 exactly
    size_t row = idx / 160;
    int    c   = (int)(idx % 160);
    dst[idx] = (c < AFEAT) ? fa[row * AFEAT + c] : 0.0f;
}

// ---------------------------------------------------------------------------
// Gather-sum for message loop: out[N x 288]
//   cols [0,256)  = sum_k msg[a2a[row,k]]
//   cols [256,270)= sum_k f_bonds[a2b[row,k]]
//   cols [270,288)= 0
// One block (288 threads = 9 waves) per atom row. msg fits in 192MB L2.
// ---------------------------------------------------------------------------
__global__ void gather_concat_kernel(const float* __restrict__ msg,
                                     const float* __restrict__ f_bonds,
                                     const int* __restrict__ a2a,
                                     const int* __restrict__ a2b,
                                     float* __restrict__ out) {
    const size_t row = blockIdx.x;
    const int c = threadIdx.x;
    float s = 0.0f;
    if (c < HH) {
        #pragma unroll
        for (int k = 0; k < KNEI; ++k) {
            size_t idx = (size_t)a2a[row * KNEI + k];
            s += msg[idx * HH + c];
        }
    } else {
        int j = c - HH;
        if (j < BFEAT) {
            #pragma unroll
            for (int k = 0; k < KNEI; ++k) {
                size_t idx = (size_t)a2b[row * KNEI + k];
                s += f_bonds[idx * BFEAT + j];
            }
        }
    }
    out[row * 288 + c] = s;
}

// Final gather + LayerNorm + PReLU: out[N x 256] written with row stride 288.
__global__ void gather_ln_kernel(const float* __restrict__ msg,
                                 const int* __restrict__ a2a,
                                 const float* __restrict__ g,
                                 const float* __restrict__ b,
                                 const float* __restrict__ alpha,
                                 float* __restrict__ out) {
    __shared__ float sh[HH];
    __shared__ float sh2[HH];
    const size_t row = blockIdx.x;
    const int c = threadIdx.x;
    float s = 0.0f;
    #pragma unroll
    for (int k = 0; k < KNEI; ++k) {
        size_t idx = (size_t)a2a[row * KNEI + k];
        s += msg[idx * HH + c];
    }
    sh[c] = s; sh2[c] = s * s;
    __syncthreads();
    for (int off = 128; off > 0; off >>= 1) {
        if (c < off) { sh[c] += sh[c + off]; sh2[c] += sh2[c + off]; }
        __syncthreads();
    }
    float mu  = sh[0] * (1.0f / HH);
    float var = sh2[0] * (1.0f / HH) - mu * mu;
    float rs  = rsqrtf(var + 1e-5f);
    float y = (s - mu) * rs * g[c] + b[c];
    float al = *alpha;
    y = (y >= 0.0f) ? y : al * y;
    out[row * 288 + c] = y;
}

// ---------------------------------------------------------------------------
// WMMA GEMM: out[M x 256] = epi( A[M x K] @ Bpk + bias )
//   - 8 waves/block: wave (w>>2) owns M rows [mrow,mrow+16), (w&3) owns 64 cols
//   - block owns 32 full 256-wide rows -> LayerNorm fused via LDS
//   - A chunks kc < split_kc read from A (stride lda), else A2 (stride lda2)
//   - DO_LN: bias + LayerNorm(gamma,beta) + PReLU(*alpha)
//   - !DO_LN: bias (+ resid) (+ zero row 0)
// A fragment layout (16-bit A 16x32): lane m = lane%16,
//   elems 0..7 -> K = kb + e + (laneHi?8:0), elems 8..15 -> K = kb+16+(e-8)+(laneHi?8:0)
// ---------------------------------------------------------------------------
template <bool DO_LN>
__global__ __launch_bounds__(256)
void gemm_wmma_kernel(const float* __restrict__ A,  int lda,
                      const float* __restrict__ A2, int lda2, int split_kc, int nkc,
                      const unsigned short* __restrict__ Bpk,
                      const float* __restrict__ bias,
                      const float* __restrict__ resid, int zero_row0,
                      const float* __restrict__ gamma, const float* __restrict__ beta,
                      const float* __restrict__ alpha,
                      float* __restrict__ out, int ldo) {
    const int tid    = threadIdx.x;
    const int wave   = tid >> 5;
    const int lane   = tid & 31;
    const int laneHi = (lane >> 4) & 1;
    const int lm     = lane & 15;
    const int mrow   = (wave >> 2) << 4;     // 0 or 16
    const int ncol   = (wave & 3) << 6;      // 0,64,128,192
    const size_t tileM = (size_t)blockIdx.x * 32;

    v8f acc[4] = {};

    const size_t mA = tileM + mrow + lm;     // A row for this lane (both halves)
    const int ntBase = ncol >> 4;

    for (int kc = 0; kc < nkc; ++kc) {
        const float* src; int kb, stride;
        if (kc < split_kc) { src = A;  kb = kc * 32;              stride = lda;  }
        else               { src = A2; kb = (kc - split_kc) * 32; stride = lda2; }
        const float* arow = src + mA * (size_t)stride + kb + (laneHi ? 8 : 0);
        float4 q0 = *(const float4*)(arow + 0);
        float4 q1 = *(const float4*)(arow + 4);
        float4 q2 = *(const float4*)(arow + 16);
        float4 q3 = *(const float4*)(arow + 20);
        v8u au;
        au[0] = pk2(q0.x, q0.y); au[1] = pk2(q0.z, q0.w);
        au[2] = pk2(q1.x, q1.y); au[3] = pk2(q1.z, q1.w);
        au[4] = pk2(q2.x, q2.y); au[5] = pk2(q2.z, q2.w);
        au[6] = pk2(q3.x, q3.y); au[7] = pk2(q3.z, q3.w);
        v16bf af = __builtin_bit_cast(v16bf, au);
        #pragma unroll
        for (int j = 0; j < 4; ++j) {
            const v8u bu = *(const v8u*)(Bpk +
                (((size_t)(ntBase + j) * nkc + kc) * 32 + lane) * 16);
            v16bf bf = __builtin_bit_cast(v16bf, bu);
            acc[j] = __builtin_amdgcn_wmma_f32_16x16x32_bf16(
                false, af, false, bf, (short)0, acc[j], false, false);
        }
    }

    if constexpr (!DO_LN) {
        // C/D layout: VGPR r -> M = r + (laneHi?8:0), N = ncol + j*16 + lane%16
        #pragma unroll
        for (int j = 0; j < 4; ++j) {
            #pragma unroll
            for (int r = 0; r < 8; ++r) {
                int rowl = mrow + r + (laneHi ? 8 : 0);
                int col  = ncol + j * 16 + lm;
                size_t m = tileM + rowl;
                float v = acc[j][r] + bias[col];
                if (resid) v += resid[m * (size_t)ldo + col];
                if (zero_row0 && m == 0) v = 0.0f;
                out[m * (size_t)ldo + col] = v;
            }
        }
    } else {
        __shared__ float tile[32 * 260];     // padded row stride vs bank conflicts
        __shared__ float ps[256];
        __shared__ float ps2[256];
        #pragma unroll
        for (int j = 0; j < 4; ++j) {
            #pragma unroll
            for (int r = 0; r < 8; ++r) {
                int rowl = mrow + r + (laneHi ? 8 : 0);
                int col  = ncol + j * 16 + lm;
                tile[rowl * 260 + col] = acc[j][r];
            }
        }
        __syncthreads();
        const int row = tid >> 3;            // 0..31
        const int sub = tid & 7;             // 8 threads per row
        const int c0  = sub * 32;
        float s = 0.0f, s2 = 0.0f;
        #pragma unroll 8
        for (int i = 0; i < 32; ++i) {
            int c = c0 + i;
            float v = tile[row * 260 + c] + bias[c];
            tile[row * 260 + c] = v;
            s += v; s2 += v * v;
        }
        ps[tid] = s; ps2[tid] = s2;
        __syncthreads();
        float sum = 0.0f, sq = 0.0f;
        #pragma unroll
        for (int i = 0; i < 8; ++i) { sum += ps[row * 8 + i]; sq += ps2[row * 8 + i]; }
        float mu  = sum * (1.0f / HH);
        float var = sq  * (1.0f / HH) - mu * mu;
        float rs  = rsqrtf(var + 1e-5f);
        float al  = *alpha;
        size_t m  = tileM + row;
        #pragma unroll 8
        for (int i = 0; i < 32; ++i) {
            int c = c0 + i;
            float v = tile[row * 260 + c];
            float y = (v - mu) * rs * gamma[c] + beta[c];
            y = (y >= 0.0f) ? y : al * y;
            out[m * (size_t)ldo + c] = y;
        }
    }
}

// ---------------------------------------------------------------------------
extern "C" void kernel_launch(void* const* d_in, const int* in_sizes, int n_in,
                              void* d_out, int out_size, void* d_ws, size_t ws_size,
                              hipStream_t stream) {
    (void)in_sizes; (void)n_in; (void)out_size; (void)ws_size;

    const float* f_atoms = (const float*)d_in[0];
    const float* f_bonds = (const float*)d_in[1];
    const float* wi_w    = (const float*)d_in[2];
    const float* wi_b    = (const float*)d_in[3];
    const float* wh_l0_w = (const float*)d_in[4];
    const float* wh_l0_b = (const float*)d_in[5];
    const float* wh_g    = (const float*)d_in[6];
    const float* wh_bt   = (const float*)d_in[7];
    const float* wh_a    = (const float*)d_in[8];
    const float* wh_l1_w = (const float*)d_in[9];
    const float* wh_l1_b = (const float*)d_in[10];
    const float* tune_g  = (const float*)d_in[11];
    const float* tune_b  = (const float*)d_in[12];
    const float* tune_a  = (const float*)d_in[13];
    const float* wah0_w  = (const float*)d_in[14];
    const float* wah0_b  = (const float*)d_in[15];
    const float* wah0_g  = (const float*)d_in[16];
    const float* wah0_bt = (const float*)d_in[17];
    const float* wah0_a  = (const float*)d_in[18];
    const float* wah_w   = (const float*)d_in[19];
    const float* wah_b   = (const float*)d_in[20];
    const float* wah_g   = (const float*)d_in[21];
    const float* wah_bt  = (const float*)d_in[22];
    const float* wah_a   = (const float*)d_in[23];
    const float* wo_w    = (const float*)d_in[24];
    const float* wo_b    = (const float*)d_in[25];
    const float* wo_g    = (const float*)d_in[26];
    const float* wo_bt   = (const float*)d_in[27];
    const float* wo_a    = (const float*)d_in[28];
    const int*   a2a     = (const int*)d_in[29];
    const int*   a2b     = (const int*)d_in[30];
    float*       outp    = (float*)d_out;

    // ---- carve workspace (256B-aligned slices) ----
    char* wsb = (char*)d_ws;
    size_t off = 0;
    auto take = [&](size_t bytes) -> void* {
        void* p = wsb + off;
        off += (bytes + 255) & ~(size_t)255;
        return p;
    };
    float* buf0 = (float*)take((size_t)NN * 160 * 4);   // padded f_atoms
    float* msg  = (float*)take((size_t)NN * HH  * 4);   // message state / cc scratch
    float* buf1 = (float*)take((size_t)NN * 288 * 4);   // gather concat / a_message
    float* buf2 = (float*)take((size_t)NN * HH  * 4);   // GEMM ping buffer
    unsigned short* wpk = (unsigned short*)take(2976ull * HH * 2);

    unsigned short* Wi_pk   = wpk;                       // 160x256 -> 40960 elems
    unsigned short* Wl0_pk  = wpk + 40960;               // 3 x 288x256 (73728 ea)
    unsigned short* Wl1_pk  = wpk + 262144;              // 3 x 256x256 (65536 ea)
    unsigned short* Wah0_pk = wpk + 458752;              // 160x256
    unsigned short* Wah_pk  = wpk + 499712;              // 2 x 256x256
    unsigned short* Wo_pk   = wpk + 630784;              // 512x256

    // ---- pack weights to bf16 fragment order (runs every call; tiny) ----
    pack_w_kernel<<<5 * 32, 256, 0, stream>>>(wi_w, AFEAT, 5, Wi_pk);
    for (int d = 0; d < DEPTH; ++d)
        pack_w_kernel<<<9 * 32, 256, 0, stream>>>(wh_l0_w + (size_t)d * 270 * HH, 270, 9,
                                                  Wl0_pk + (size_t)d * 73728);
    for (int d = 0; d < DEPTH; ++d)
        pack_w_kernel<<<8 * 32, 256, 0, stream>>>(wh_l1_w + (size_t)d * HH * HH, HH, 8,
                                                  Wl1_pk + (size_t)d * 65536);
    pack_w_kernel<<<5 * 32, 256, 0, stream>>>(wah0_w, AFEAT, 5, Wah0_pk);
    for (int i = 0; i < DEPTH - 1; ++i)
        pack_w_kernel<<<8 * 32, 256, 0, stream>>>(wah_w + (size_t)i * HH * HH, HH, 8,
                                                  Wah_pk + (size_t)i * 65536);
    pack_w_kernel<<<16 * 32, 256, 0, stream>>>(wo_w, 2 * HH, 16, Wo_pk);

    // ---- pad atom features ----
    pad_atoms_kernel<<<(NN * 160) / 256, 256, 0, stream>>>(f_atoms, buf0);

    const int MG = NN / 32;  // 3125 blocks, M divides exactly

    // ---- h = f_atoms @ wi_w + b, zero row 0 -> msg ----
    gemm_wmma_kernel<false><<<MG, 256, 0, stream>>>(
        buf0, 160, buf0, 160, 5, 5, Wi_pk, wi_b,
        nullptr, 1, nullptr, nullptr, nullptr, msg, HH);

    // ---- message passing loop ----
    for (int d = 0; d < DEPTH; ++d) {
        gather_concat_kernel<<<NN, 288, 0, stream>>>(msg, f_bonds, a2a, a2b, buf1);
        gemm_wmma_kernel<true><<<MG, 256, 0, stream>>>(
            buf1, 288, buf1, 288, 9, 9, Wl0_pk + (size_t)d * 73728,
            wh_l0_b + (size_t)d * HH, nullptr, 0,
            wh_g + (size_t)d * HH, wh_bt + (size_t)d * HH, wh_a + d, buf2, HH);
        gemm_wmma_kernel<false><<<MG, 256, 0, stream>>>(
            buf2, HH, buf2, HH, 8, 8, Wl1_pk + (size_t)d * 65536,
            wh_l1_b + (size_t)d * HH, msg, 1, nullptr, nullptr, nullptr, msg, HH);
    }

    // ---- a_message = prelu(ln(gather-sum msg)) -> buf1 (stride 288) ----
    gather_ln_kernel<<<NN, HH, 0, stream>>>(msg, a2a, tune_g, tune_b, tune_a, buf1);

    // ---- atom path: cc chain ----
    gemm_wmma_kernel<true><<<MG, 256, 0, stream>>>(
        buf0, 160, buf0, 160, 5, 5, Wah0_pk, wah0_b,
        nullptr, 0, wah0_g, wah0_bt, wah0_a, buf2, HH);
    gemm_wmma_kernel<true><<<MG, 256, 0, stream>>>(
        buf2, HH, buf2, HH, 8, 8, Wah_pk, wah_b,
        nullptr, 0, wah_g, wah_bt, wah_a, msg, HH);
    gemm_wmma_kernel<true><<<MG, 256, 0, stream>>>(
        msg, HH, msg, HH, 8, 8, Wah_pk + 65536, wah_b + HH,
        nullptr, 0, wah_g + HH, wah_bt + HH, wah_a + 1, buf2, HH);

    // ---- out = prelu(ln(concat(cc, a_message) @ wo_w + b)) ----
    // A chunks 0..7 from cc (buf2, stride 256), chunks 8..15 from a_message (buf1, stride 288)
    gemm_wmma_kernel<true><<<MG, 256, 0, stream>>>(
        buf2, HH, buf1, 288, 8, 16, Wo_pk, wo_b,
        nullptr, 0, wo_g, wo_bt, wo_a, outp, HH);
}
